// RecurrentGCN_TGCN_15693810499718
// MI455X (gfx1250) — matrix operand order repaired and checked by reading, not compiled, not run on hardware
//
#include <hip/hip_runtime.h>
#include <hip/hip_bf16.h>

typedef __attribute__((ext_vector_type(16))) _Float16 v16h;
typedef __attribute__((ext_vector_type(8)))  float    v8f;

#define F_IN 128
#define HDIM 32
#define GCOL 96          // 3 gates * 32
#define LDS_STRIDE 136   // 128 + 8 halves padding

// ---------------- deg / norm ----------------
__global__ void __launch_bounds__(256) k_deg_init(float* __restrict__ deg, int n) {
    int i = blockIdx.x * 256 + threadIdx.x;
    if (i < n) deg[i] = 1.0f;               // self-loop weight
}

__global__ void __launch_bounds__(256) k_deg_acc(const long long* __restrict__ ei,
                                                 const float* __restrict__ ew,
                                                 float* __restrict__ deg, long long E) {
    long long e = (long long)blockIdx.x * 256 + threadIdx.x;
    if (e < E) atomicAdd(&deg[(size_t)ei[E + e]], ew[e]);
}

__global__ void __launch_bounds__(256) k_dinv(float* __restrict__ deg, int n) {
    int i = blockIdx.x * 256 + threadIdx.x;
    if (i < n) deg[i] = rsqrtf(deg[i]);     // deg >= 1 always (self loops)
}

// ---------------- weight fragment prep (B layout for v_wmma_f32_16x16x32_f16) ------
// Wfrag[((s*6+t)*32 + L)*16 + e] = Wcat[k][col], k = s*32 + (L>>4)*16 + e,
// col = t*16 + (L&15);  Wcat = [Wc_z | Wc_r | Wc_h]  (128 x 96)
__global__ void __launch_bounds__(256) k_prep_wfrag(const float* __restrict__ wcz,
                                                    const float* __restrict__ wcr,
                                                    const float* __restrict__ wch,
                                                    _Float16* __restrict__ wfrag) {
    int idx = blockIdx.x * 256 + threadIdx.x;
    if (idx >= 4 * 6 * 32 * 16) return;
    int e  = idx & 15;
    int L  = (idx >> 4) & 31;
    int st = idx >> 9;          // s*6 + t
    int t  = st % 6;
    int s  = st / 6;
    int k   = s * 32 + (L >> 4) * 16 + e;
    int col = t * 16 + (L & 15);
    const float* w = (col < 32) ? wcz : ((col < 64) ? wcr : wch);
    int cc = col & 31;
    wfrag[idx] = (_Float16)w[k * 32 + cc];
}

// ---------------- XW = x @ Wcat  (WMMA f16 -> f32) ----------------
// Block = 256 threads = 8 waves; block covers 128 rows; each wave: 16 rows x 96 cols.
__global__ void __launch_bounds__(256) k_gemm_xw(const float* __restrict__ x,
                                                 const _Float16* __restrict__ wfrag,
                                                 float* __restrict__ xw, int n) {
    __shared__ _Float16 lx[128 * LDS_STRIDE];
    const int tid = threadIdx.x;
    const int blockRow = blockIdx.x * 128;

    // stage 128x128 fp32 tile -> f16 LDS (zero-padded past n)
    for (int idx = tid; idx < 128 * 128; idx += 256) {
        int r = idx >> 7, c = idx & 127;
        int gr = blockRow + r;
        float v = (gr < n) ? x[(size_t)gr * F_IN + c] : 0.0f;
        lx[r * LDS_STRIDE + c] = (_Float16)v;
    }
    __syncthreads();

    const int wave  = tid >> 5;
    const int lane  = tid & 31;
    const int m16   = lane & 15;
    const int rhalf = lane >> 4;
    const int rowBase = wave * 16;

    v8f acc[6];
    #pragma unroll
    for (int t = 0; t < 6; ++t)
        #pragma unroll
        for (int j = 0; j < 8; ++j) acc[t][j] = 0.0f;

    #pragma unroll
    for (int s = 0; s < 4; ++s) {
        // A fragment: lane holds row m16; halves 0..15 -> K=rhalf*16+0..15
        v16h a;
        #pragma unroll
        for (int e = 0; e < 16; ++e) {
            int k = s * 32 + rhalf * 16 + e;
            a[e] = lx[(rowBase + m16) * LDS_STRIDE + k];
        }
        #pragma unroll
        for (int t = 0; t < 6; ++t) {
            v16h b = *(const v16h*)(wfrag + ((size_t)(s * 6 + t) * 32 + lane) * 16);
            acc[t] = __builtin_amdgcn_wmma_f32_16x16x32_f16(
                false, a, false, b, (short)0, acc[t], false, false);
        }
    }

    // D: lane holds col m16; VGPR v -> row v + rhalf*8
    #pragma unroll
    for (int t = 0; t < 6; ++t) {
        #pragma unroll
        for (int v = 0; v < 8; ++v) {
            int grow = blockRow + rowBase + v + rhalf * 8;
            if (grow < n) xw[(size_t)grow * GCOL + t * 16 + m16] = acc[t][v];
        }
    }
}

// ---------------- agg init with self-loop term: agg = XW * dinv^2 ----------------
__global__ void __launch_bounds__(256) k_agg_init(const float* __restrict__ xw,
                                                  const float* __restrict__ dinv,
                                                  float* __restrict__ agg, int n) {
    int idx = blockIdx.x * 256 + threadIdx.x;
    if (idx < n * GCOL) {
        float d = dinv[idx / GCOL];
        agg[idx] = xw[idx] * d * d;
    }
}

// ---------------- edge scatter: one wave per edge, 3 features per lane ----------------
__global__ void __launch_bounds__(256) k_edge_scatter(const long long* __restrict__ ei,
                                                      const float* __restrict__ ew,
                                                      const float* __restrict__ dinv,
                                                      const float* __restrict__ xw,
                                                      float* __restrict__ agg, long long E) {
    long long e = ((long long)blockIdx.x * 256 + threadIdx.x) >> 5;
    int lane = threadIdx.x & 31;
    if (e >= E) return;
    long long r = ei[e];
    long long c = ei[E + e];
    float nrm = dinv[(size_t)r] * ew[e] * dinv[(size_t)c];
    const float* src = xw + (size_t)r * GCOL;
    float* dst = agg + (size_t)c * GCOL;
    #pragma unroll
    for (int p = 0; p < 3; ++p) {
        int f = lane + p * 32;
        atomicAdd(&dst[f], nrm * src[f]);
    }
}

// ---------------- gates + head: one wave32 per node ----------------
__global__ void __launch_bounds__(256) k_gates(const float* __restrict__ agg,
                                               const float* __restrict__ hprev,
                                               const float* __restrict__ bcz,
                                               const float* __restrict__ Wlz,
                                               const float* __restrict__ blz,
                                               const float* __restrict__ bcr,
                                               const float* __restrict__ Wlr,
                                               const float* __restrict__ blr,
                                               const float* __restrict__ bch,
                                               const float* __restrict__ Wlh,
                                               const float* __restrict__ blh,
                                               const float* __restrict__ whead,
                                               const float* __restrict__ bhead,
                                               float* __restrict__ out, int n) {
    __shared__ float sWz[32 * 65], sWr[32 * 65], sWh[32 * 65];
    int tid = threadIdx.x;
    for (int idx = tid; idx < 2048; idx += 256) {
        int j = idx >> 6, k = idx & 63;
        sWz[j * 65 + k] = Wlz[idx];
        sWr[j * 65 + k] = Wlr[idx];
        sWh[j * 65 + k] = Wlh[idx];
    }
    __syncthreads();

    int wave = tid >> 5, lane = tid & 31;
    int i = blockIdx.x * 8 + wave;
    if (i >= n) return;

    float cz = agg[(size_t)i * GCOL +      lane] + bcz[lane];
    float cr = agg[(size_t)i * GCOL + 32 + lane] + bcr[lane];
    float ch = agg[(size_t)i * GCOL + 64 + lane] + bch[lane];
    float hp = hprev[(size_t)i * HDIM + lane];

    const float* wz = &sWz[lane * 65];
    const float* wr = &sWr[lane * 65];
    const float* wh = &sWh[lane * 65];

    float az = blz[lane], ar = blr[lane];
    #pragma unroll
    for (int k = 0; k < 32; ++k) {
        float czk = __shfl(cz, k);
        float crk = __shfl(cr, k);
        float hpk = __shfl(hp, k);
        az += czk * wz[k] + hpk * wz[32 + k];
        ar += crk * wr[k] + hpk * wr[32 + k];
    }
    float Z = 1.0f / (1.0f + __expf(-az));
    float R = 1.0f / (1.0f + __expf(-ar));
    float hr = hp * R;

    float ah = blh[lane];
    #pragma unroll
    for (int k = 0; k < 32; ++k) {
        ah += __shfl(ch, k) * wh[k] + __shfl(hr, k) * wh[32 + k];
    }
    float Ht = tanhf(ah);
    float Hn = Z * hp + (1.0f - Z) * Ht;

    out[(size_t)n + (size_t)i * HDIM + lane] = Hn;   // Hnew

    float contrib = fmaxf(Hn, 0.0f) * whead[lane];
    #pragma unroll
    for (int off = 16; off > 0; off >>= 1) contrib += __shfl_xor(contrib, off);
    if (lane == 0) out[i] = contrib + bhead[0];      // y
}

// ---------------- launch ----------------
extern "C" void kernel_launch(void* const* d_in, const int* in_sizes, int n_in,
                              void* d_out, int out_size, void* d_ws, size_t ws_size,
                              hipStream_t stream) {
    const float*     x     = (const float*)d_in[0];
    const long long* ei    = (const long long*)d_in[1];
    const float*     ew    = (const float*)d_in[2];
    const float*     hprev = (const float*)d_in[3];
    const float* Wcz = (const float*)d_in[4];
    const float* bcz = (const float*)d_in[5];
    const float* Wlz = (const float*)d_in[6];
    const float* blz = (const float*)d_in[7];
    const float* Wcr = (const float*)d_in[8];
    const float* bcr = (const float*)d_in[9];
    const float* Wlr = (const float*)d_in[10];
    const float* blr = (const float*)d_in[11];
    const float* Wch = (const float*)d_in[12];
    const float* bch = (const float*)d_in[13];
    const float* Wlh = (const float*)d_in[14];
    const float* blh = (const float*)d_in[15];
    const float* whd = (const float*)d_in[16];
    const float* bhd = (const float*)d_in[17];
    float* out = (float*)d_out;

    const int       N = in_sizes[0] / F_IN;           // 100000
    const long long E = (long long)in_sizes[2];       // 3200000

    // workspace layout (bytes)
    char* ws = (char*)d_ws;
    float*     deg   = (float*)(ws + 0);                       // N*4        = 400000
    _Float16*  wfrag = (_Float16*)(ws + 401408);               // 12288*2    = 24576
    float*     xw    = (float*)(ws + 425984);                  // N*96*4     = 38400000
    float*     agg   = (float*)(ws + 425984 + 38400000ULL);    // N*96*4

    const int nb256 = (N + 255) / 256;

    k_deg_init<<<nb256, 256, 0, stream>>>(deg, N);
    k_deg_acc<<<(int)((E + 255) / 256), 256, 0, stream>>>(ei, ew, deg, E);
    k_dinv<<<nb256, 256, 0, stream>>>(deg, N);

    k_prep_wfrag<<<48, 256, 0, stream>>>(Wcz, Wcr, Wch, wfrag);

    k_gemm_xw<<<(N + 127) / 128, 256, 0, stream>>>(x, wfrag, xw, N);

    k_agg_init<<<(N * GCOL + 255) / 256, 256, 0, stream>>>(xw, deg, agg, N);

    long long scatterThreads = E * 32;
    k_edge_scatter<<<(int)((scatterThreads + 255) / 256), 256, 0, stream>>>(
        ei, ew, deg, xw, agg, E);

    k_gates<<<(N + 7) / 8, 256, 0, stream>>>(agg, hprev,
        bcz, Wlz, blz, bcr, Wlr, blr, bch, Wlh, blh, whd, bhd, out, N);
}
